// QuantumAttention_51144470560878
// MI455X (gfx1250) — compile-verified, matrix operand order checked by reading, not run on hardware
//
#include <hip/hip_runtime.h>

typedef __attribute__((ext_vector_type(16))) _Float16 v16h;
typedef __attribute__((ext_vector_type(8)))  _Float16 v8h;
typedef __attribute__((ext_vector_type(8)))  float    v8f;

#define B_ 4
#define L_ 2048
#define H_ 16
#define E_ 64
#define CSCALE 0.0625f          // ENTANGLEMENT_FACTOR / sqrt(E) = 0.5/8
#define LOG2E  1.44269504088896340736f

// ---------------------------------------------------------------------------
// Phase 1: row sums  ks[b,h,s] = sum_e keys[b,s,h,e], vs[b,h,l] = sum_d values
// flat thread idx == (b*L + l)*H + h == row index into [b,l,h,:] (64 floats)
// ---------------------------------------------------------------------------
__global__ __launch_bounds__(256)
void qa_rowsum(const float* __restrict__ keys, const float* __restrict__ values,
               float* __restrict__ ks, float* __restrict__ vs) {
    int idx = blockIdx.x * 256 + threadIdx.x;          // 0 .. B*L*H-1
    int h  = idx % H_;
    int bl = idx / H_;
    int l  = bl % L_;
    int b  = bl / L_;
    const float4* kp = (const float4*)(keys   + (size_t)idx * E_);
    const float4* vp = (const float4*)(values + (size_t)idx * E_);
    float sk = 0.f, sv = 0.f;
#pragma unroll
    for (int i = 0; i < E_ / 4; ++i) {
        float4 a = kp[i]; float4 c = vp[i];
        sk += (a.x + a.y) + (a.z + a.w);
        sv += (c.x + c.y) + (c.z + c.w);
    }
    int o = (b * H_ + h) * L_ + l;
    ks[o] = sk;
    vs[o] = sv;
}

// ---------------------------------------------------------------------------
// Phase 2: per-(b,h) inclusive prefix max/min over ks (L=2048, 256 thr x 8)
// ---------------------------------------------------------------------------
__global__ __launch_bounds__(256)
void qa_scan(const float* __restrict__ ks,
             float* __restrict__ pmax, float* __restrict__ pmin) {
    __shared__ float smax[256];
    __shared__ float smin[256];
    int bh = blockIdx.x;
    const float* src = ks + (size_t)bh * L_;
    int t = threadIdx.x;
    float lx[8], ln[8];
    float mx = -__builtin_inff(), mn = __builtin_inff();
#pragma unroll
    for (int j = 0; j < 8; ++j) {
        float v = src[t * 8 + j];
        mx = fmaxf(mx, v);  mn = fminf(mn, v);
        lx[j] = mx;         ln[j] = mn;
    }
    smax[t] = mx; smin[t] = mn;
    __syncthreads();
    for (int off = 1; off < 256; off <<= 1) {
        float ta = (t >= off) ? smax[t - off] : -__builtin_inff();
        float tb = (t >= off) ? smin[t - off] :  __builtin_inff();
        __syncthreads();
        smax[t] = fmaxf(smax[t], ta);
        smin[t] = fminf(smin[t], tb);
        __syncthreads();
    }
    float ox = (t > 0) ? smax[t - 1] : -__builtin_inff();
    float on = (t > 0) ? smin[t - 1] :  __builtin_inff();
    float* dx = pmax + (size_t)bh * L_;
    float* dn = pmin + (size_t)bh * L_;
#pragma unroll
    for (int j = 0; j < 8; ++j) {
        dx[t * 8 + j] = fmaxf(lx[j], ox);
        dn[t * 8 + j] = fminf(ln[j], on);
    }
}

// ---------------------------------------------------------------------------
// Phase 3: values[b,s,h,d] f32 -> Vt[b,h,d,s] f16 (transpose via LDS tile)
// ---------------------------------------------------------------------------
__global__ __launch_bounds__(256)
void qa_transpose(const float* __restrict__ values, _Float16* __restrict__ Vt) {
    __shared__ float tile[32][E_ + 1];
    int s0 = blockIdx.x * 32, h = blockIdx.y, b = blockIdx.z;
    int t = threadIdx.x;
#pragma unroll
    for (int i = 0; i < 8; ++i) {
        int flat = t + i * 256;
        int s = flat >> 6, d = flat & 63;
        tile[s][d] = values[(((size_t)b * L_ + (s0 + s)) * H_ + h) * E_ + d];
    }
    __syncthreads();
    int d  = t >> 2;
    int sc = (t & 3) * 8;
    v8h o;
#pragma unroll
    for (int j = 0; j < 8; ++j) o[j] = (_Float16)tile[sc + j][d];
    *(v8h*)(Vt + (((size_t)(b * H_ + h)) * E_ + d) * L_ + s0 + sc) = o;
}

// ---------------------------------------------------------------------------
// Phase 4: flash-style causal P@V with v_wmma_f32_16x16x32_f16
// 1 wave handles a 16-row query tile; 4 waves/block share the ks row in LDS.
// ---------------------------------------------------------------------------
__global__ __launch_bounds__(128)
void qa_flash(const float* __restrict__ vs, const float* __restrict__ ksg,
              const float* __restrict__ pmax, const float* __restrict__ pmin,
              const _Float16* __restrict__ Vt, float* __restrict__ out) {
    __shared__ float ks_sh[L_];
    int b = blockIdx.z, h = blockIdx.y;
    int bh = b * H_ + h;

    // cooperative load of ksum row (8 KB) into LDS
    const float4* src4 = (const float4*)(ksg + (size_t)bh * L_);
    float4* sh4 = (float4*)ks_sh;
#pragma unroll
    for (int i = 0; i < (L_ / 4) / 128; ++i)
        sh4[threadIdx.x + i * 128] = src4[threadIdx.x + i * 128];
    __syncthreads();

    int wave = threadIdx.x >> 5;
    int lane = threadIdx.x & 31;
    int row  = lane & 15;
    int hi   = lane >> 4;                       // 0 or 1 (lane half)
    int l0   = (blockIdx.x * 4 + wave) * 16;
    int l    = l0 + row;

    // fold softmax scale AND log2(e) into the per-row coefficient so the
    // inner loop is exactly one v_fma + one v_exp per P element.
    float a2 = (CSCALE * LOG2E) * vs[(size_t)bh * L_ + l];
    float pm = pmax[(size_t)bh * L_ + l];
    float pn = pmin[(size_t)bh * L_ + l];
    float m2 = (a2 >= 0.f) ? a2 * pm : a2 * pn;       // exact row max (s<=l), log2 domain

    v8f acc[4];
#pragma unroll
    for (int g = 0; g < 4; ++g) acc[g] = (v8f){};
    float zpart = 0.f;
    int kb   = hi ? 8 : 0;                            // lane-half K offset
    int dcol = lane & 15;
    const _Float16* vtb = Vt + (size_t)bh * E_ * L_;

    for (int s0 = 0; s0 < l0 + 16; s0 += 32) {
        bool full = (s0 + 31) <= l0;                  // whole tile unmasked
        float p[16];
#pragma unroll
        for (int j = 0; j < 8; ++j) {
            int s = s0 + kb + j;
            float e = __builtin_amdgcn_exp2f(__builtin_fmaf(a2, ks_sh[s], -m2));
            p[j] = (full || s <= l) ? e : 0.f;
        }
#pragma unroll
        for (int j = 0; j < 8; ++j) {
            int s = s0 + 16 + kb + j;
            float e = __builtin_amdgcn_exp2f(__builtin_fmaf(a2, ks_sh[s], -m2));
            p[8 + j] = (full || s <= l) ? e : 0.f;
        }
        // tree-structured row-sum accumulation (depth 4, not a 16-deep chain)
        {
            float t0 = (p[0]  + p[1])  + (p[2]  + p[3]);
            float t1 = (p[4]  + p[5])  + (p[6]  + p[7]);
            float t2 = (p[8]  + p[9])  + (p[10] + p[11]);
            float t3 = (p[12] + p[13]) + (p[14] + p[15]);
            zpart += (t0 + t1) + (t2 + t3);
        }

        // A fragment: 16-bit A 16x32 layout (lanes 0-15: K=0-7,16-23; hi half +8)
        v16h afrag;
#pragma unroll
        for (int j = 0; j < 16; ++j) afrag[j] = (_Float16)p[j];

        int soff0 = s0 + (hi ? 8 : 0);
#pragma unroll
        for (int g = 0; g < 4; ++g) {
            const _Float16* col = vtb + (size_t)(g * 16 + dcol) * L_;
            v8h blo = *(const v8h*)(col + soff0);         // K = soff0..+7
            v8h bhi = *(const v8h*)(col + soff0 + 16);    // K = soff0+16..+23
            v16h bfrag = __builtin_shufflevector(blo, bhi,
                0, 1, 2, 3, 4, 5, 6, 7, 8, 9, 10, 11, 12, 13, 14, 15);
            acc[g] = __builtin_amdgcn_wmma_f32_16x16x32_f16(
                false, afrag, false, bfrag, (short)0, acc[g], false, false);
        }
    }

    // full row sums: lane r and lane r+16 hold complementary K halves of row r
    float zrow = zpart + __shfl_xor(zpart, 16, 32);

    // C/D layout: lanes 0-15 hold M=0..7 (VGPR j = M), lanes 16-31 hold M=8..15
    int mbase = hi ? 8 : 0;
    float rz[8];
#pragma unroll
    for (int j = 0; j < 8; ++j)
        rz[j] = 1.0f / __shfl(zrow, mbase + j, 32);

#pragma unroll
    for (int j = 0; j < 8; ++j) {
        size_t obase = (((size_t)b * L_ + (l0 + mbase + j)) * H_ + h) * E_ + dcol;
        out[obase +  0] = acc[0][j] * rz[j];
        out[obase + 16] = acc[1][j] * rz[j];
        out[obase + 32] = acc[2][j] * rz[j];
        out[obase + 48] = acc[3][j] * rz[j];
    }
}

// ---------------------------------------------------------------------------
extern "C" void kernel_launch(void* const* d_in, const int* in_sizes, int n_in,
                              void* d_out, int out_size, void* d_ws, size_t ws_size,
                              hipStream_t stream) {
    // inputs: 0=queries (unused), 1=keys, 2=values, 3=params (cancels in softmax)
    const float* keys   = (const float*)d_in[1];
    const float* values = (const float*)d_in[2];
    float* out = (float*)d_out;

    char* ws = (char*)d_ws;
    const size_t N = (size_t)B_ * H_ * L_;            // 131072
    float*    vs   = (float*)(ws);
    float*    ks   = (float*)(ws + 1 * N * 4);
    float*    pmax = (float*)(ws + 2 * N * 4);
    float*    pmin = (float*)(ws + 3 * N * 4);
    _Float16* Vt   = (_Float16*)(ws + 4 * N * 4);     // B*H*E*L f16 = 16 MiB

    qa_rowsum  <<<dim3((unsigned)(N / 256)), dim3(256), 0, stream>>>(keys, values, ks, vs);
    qa_scan    <<<dim3(B_ * H_),             dim3(256), 0, stream>>>(ks, pmax, pmin);
    qa_transpose<<<dim3(L_ / 32, H_, B_),    dim3(256), 0, stream>>>(values, Vt);
    qa_flash   <<<dim3(L_ / 64, H_, B_),     dim3(128), 0, stream>>>(vs, ks, pmax, pmin, Vt, out);
}
